// ScaledDotProductAttentionLayer_47193100648748
// MI455X (gfx1250) — matrix operand order repaired and checked by reading, not compiled, not run on hardware
//
#include <hip/hip_runtime.h>
#include <hip/hip_bf16.h>

typedef _Float16 v16h __attribute__((ext_vector_type(16)));
typedef _Float16 v8h  __attribute__((ext_vector_type(8)));
typedef _Float16 v4h  __attribute__((ext_vector_type(4)));
typedef float    v8f  __attribute__((ext_vector_type(8)));
typedef float    v4f  __attribute__((ext_vector_type(4)));

#define B_        8
#define L_        2048
#define D_        1024
#define QT        32      // q rows per block
#define KT        64      // k cols per tile
#define NTHREADS  256     // 8 waves (wave32)
#define QSTRIDE   1032    // halves, padded: 2064B -> 4-bank shift per row
#define VSTRIDE   72      // halves, padded: 144B  -> 36-bank shift per row
#define PSTRIDE   72
#define NEG_INF_F (-2147483648.0f)  // -2^31, matches reference
#define SCALE_F   (0.03125f)        // 1/sqrt(1024)

struct SMem {
  _Float16 Q[QT * QSTRIDE];   // 66048 B  : q tile, f16
  _Float16 Vt[D_ * VSTRIDE];  // 147456 B : V tile transposed [d][k], f16
  float    S[QT * KT];        // 8192 B   : raw masked/scaled scores
  _Float16 P[QT * PSTRIDE];   // 4608 B   : probabilities, f16
  float    red[QT * 4];       // 512 B    : row-segment reduction scratch
  float    m_run[QT];
  float    l_run[QT];
  float    inv_l[QT];
};                            // ~227 KB total (<320 KB WGP LDS)

// A fragment (16x32 f16): lane m=l%16, h=l/16; element e -> K = e + 16h (contiguous run)
__device__ __forceinline__ v16h load_a_frag_lds(const _Float16* row, int d0, int h) {
  v8h lo = *(const v8h*)(row + d0 + 16 * h);
  v8h hi = *(const v8h*)(row + d0 + 16 * h + 8);
  v16h r;
#pragma unroll
  for (int i = 0; i < 8; ++i) { r[i] = lo[i]; r[i + 8] = hi[i]; }
  return r;
}

// B fragment (32x16 f16): lane n=l%16, h=l/16; element e -> K = e + 16h (contiguous run)
__device__ __forceinline__ v16h load_b_frag_lds(const _Float16* row, int k0, int h) {
  v8h lo = *(const v8h*)(row + k0 + 16 * h);
  v8h hi = *(const v8h*)(row + k0 + 16 * h + 8);
  v16h r;
#pragma unroll
  for (int i = 0; i < 8; ++i) { r[i] = lo[i]; r[i + 8] = hi[i]; }
  return r;
}

// B fragment from global f32 (K matrix row): 16 consecutive f32 starting at p
__device__ __forceinline__ v16h load_b_frag_gf32(const float* p) {
  v4f f0 = *(const v4f*)(p);
  v4f f1 = *(const v4f*)(p + 4);
  v4f f2 = *(const v4f*)(p + 8);
  v4f f3 = *(const v4f*)(p + 12);
  v16h r;
#pragma unroll
  for (int i = 0; i < 4; ++i) {
    r[i]      = (_Float16)f0[i];
    r[i + 4]  = (_Float16)f1[i];
    r[i + 8]  = (_Float16)f2[i];
    r[i + 12] = (_Float16)f3[i];
  }
  return r;
}

__global__ __launch_bounds__(NTHREADS)
void sdpa_flash_f16wmma(const float* __restrict__ Qg,
                        const float* __restrict__ Kg,
                        const float* __restrict__ Vg,
                        const unsigned char* __restrict__ qmask,
                        const unsigned char* __restrict__ kmask,
                        float* __restrict__ Og) {
  extern __shared__ char smem_raw[];
  SMem& sm = *reinterpret_cast<SMem*>(smem_raw);

  const int tid  = threadIdx.x;
  const int wave = tid >> 5;
  const int lane = tid & 31;
  const int ln16 = lane & 15;
  const int h    = lane >> 4;
  const int b    = blockIdx.y;
  const int q0   = blockIdx.x * QT;
  const long qbase = (long)(b * L_ + q0) * D_;

  // -------- stage Q tile (f32 -> f16 LDS) --------
#pragma unroll 4
  for (int row = 0; row < QT; ++row) {
    v4f f = *(const v4f*)(Qg + qbase + (long)row * D_ + tid * 4);
    v4h hh;
#pragma unroll
    for (int i = 0; i < 4; ++i) hh[i] = (_Float16)f[i];
    *(v4h*)(&sm.Q[row * QSTRIDE + tid * 4]) = hh;
  }
  if (tid < QT) { sm.m_run[tid] = NEG_INF_F; sm.l_run[tid] = 0.0f; }
  __syncthreads();

  // QK wave roles: 2x4 grid of 16x16 S sub-tiles
  const int mt = wave >> 2;   // 0..1
  const int nt = wave & 3;    // 0..3
  const _Float16* qrow = &sm.Q[(16 * mt + ln16) * QSTRIDE];

  // compute one S tile (QT x KT) for k-tile at k0, write masked+scaled to LDS
  auto qk_to_S = [&](int k0) {
    const float* krow = Kg + (long)(b * L_ + k0 + 16 * nt + ln16) * D_;
    v8f acc = {};
#pragma unroll 4
    for (int ks = 0; ks < D_ / 32; ++ks) {
      const int d0 = 32 * ks;
      v16h a  = load_a_frag_lds(qrow, d0, h);
      v16h bb = load_b_frag_gf32(krow + d0 + 16 * h);
      acc = __builtin_amdgcn_wmma_f32_16x16x32_f16(false, a, false, bb,
                                                   (short)0, acc, false, false);
    }
    const int col = 16 * nt + ln16;
    const bool km = kmask[b * L_ + k0 + col] != 0;
#pragma unroll
    for (int r = 0; r < 8; ++r) {
      const int row = 16 * mt + r + 8 * h;
      sm.S[row * KT + col] = km ? acc[r] * SCALE_F : NEG_INF_F;
    }
  };

  // -------- pass 1: softmax statistics (row max m, row sum l) --------
  for (int k0 = 0; k0 < L_; k0 += KT) {
    qk_to_S(k0);
    __syncthreads();
    if (tid < 128) {                    // per-row 16-col segment max
      const int row = tid >> 2, seg = tid & 3;
      const float* sp = &sm.S[row * KT + seg * 16];
      float mx = sp[0];
#pragma unroll
      for (int j = 1; j < 16; ++j) mx = fmaxf(mx, sp[j]);
      sm.red[row * 4 + seg] = mx;
    }
    __syncthreads();
    if (tid < QT) {                     // combine + rescale l
      const float mo = sm.m_run[tid];
      float mn = fmaxf(fmaxf(sm.red[tid * 4], sm.red[tid * 4 + 1]),
                       fmaxf(sm.red[tid * 4 + 2], sm.red[tid * 4 + 3]));
      mn = fmaxf(mo, mn);
      sm.l_run[tid] *= __expf(mo - mn);
      sm.m_run[tid] = mn;
    }
    __syncthreads();
    if (tid < 128) {                    // per-row segment sum of exp
      const int row = tid >> 2, seg = tid & 3;
      const float mn = sm.m_run[row];
      const float* sp = &sm.S[row * KT + seg * 16];
      float sum = 0.0f;
#pragma unroll
      for (int j = 0; j < 16; ++j) sum += __expf(sp[j] - mn);
      sm.red[row * 4 + seg] = sum;
    }
    __syncthreads();
    if (tid < QT)
      sm.l_run[tid] += sm.red[tid * 4] + sm.red[tid * 4 + 1] +
                       sm.red[tid * 4 + 2] + sm.red[tid * 4 + 3];
    __syncthreads();
  }
  if (tid < QT) sm.inv_l[tid] = 1.0f / sm.l_run[tid];
  __syncthreads();

  // -------- pass 2: O = P * V, each wave owns a 128-wide d chunk --------
  v8f o[2][8];
#pragma unroll
  for (int m2 = 0; m2 < 2; ++m2)
#pragma unroll
    for (int nv = 0; nv < 8; ++nv) o[m2][nv] = (v8f){};

  for (int k0 = 0; k0 < L_; k0 += KT) {
    // stage V tile transposed: Vt[d][k] f16 (coalesced f32 reads, b64 LDS stores)
    {
      const int dloc = tid & 15;
      const int kk   = (tid >> 4) * 4;
      const float* vbase = Vg + (long)(b * L_ + k0 + kk) * D_;
#pragma unroll 4
      for (int db = 0; db < D_; db += 16) {
        const int d = db + dloc;
        v4h hh;
#pragma unroll
        for (int j = 0; j < 4; ++j) hh[j] = (_Float16)vbase[(long)j * D_ + d];
        *(v4h*)(&sm.Vt[d * VSTRIDE + kk]) = hh;
      }
    }
    qk_to_S(k0);
    __syncthreads();

    // P = exp(s - m) / l  (masked entries -> 0; covers all-masked rows too)
#pragma unroll
    for (int i = 0; i < 8; ++i) {
      const int idx = tid + NTHREADS * i;
      const int row = idx >> 6, col = idx & 63;
      const float s = sm.S[row * KT + col];
      const float p = (s == NEG_INF_F)
                          ? 0.0f
                          : __expf(s - sm.m_run[row]) * sm.inv_l[row];
      sm.P[row * PSTRIDE + col] = (_Float16)p;
    }
    __syncthreads();

    // O += P(32xKT) * V(KT x 128-chunk)
#pragma unroll
    for (int ks = 0; ks < KT / 32; ++ks) {
      v16h bv[8];
#pragma unroll
      for (int nv = 0; nv < 8; ++nv) {
        const int dcol = 128 * wave + 16 * nv + ln16;
        bv[nv] = load_b_frag_lds(&sm.Vt[dcol * VSTRIDE], 32 * ks, h);
      }
#pragma unroll
      for (int m2 = 0; m2 < 2; ++m2) {
        v16h a = load_a_frag_lds(&sm.P[(16 * m2 + ln16) * PSTRIDE], 32 * ks, h);
#pragma unroll
        for (int nv = 0; nv < 8; ++nv)
          o[m2][nv] = __builtin_amdgcn_wmma_f32_16x16x32_f16(
              false, a, false, bv[nv], (short)0, o[m2][nv], false, false);
      }
    }
    __syncthreads();   // before Vt/S/P are overwritten next iteration
  }

  // -------- store O (query-mask applied as 0/1 factor) --------
#pragma unroll
  for (int m2 = 0; m2 < 2; ++m2) {
#pragma unroll
    for (int r = 0; r < 8; ++r) {
      const int row = 16 * m2 + r + 8 * h;
      const float qs = qmask[b * L_ + q0 + row] ? 1.0f : 0.0f;
#pragma unroll
      for (int nv = 0; nv < 8; ++nv) {
        const int col = 128 * wave + 16 * nv + ln16;
        Og[qbase + (long)row * D_ + col] = o[m2][nv][r] * qs;
      }
    }
  }
}

extern "C" void kernel_launch(void* const* d_in, const int* in_sizes, int n_in,
                              void* d_out, int out_size, void* d_ws, size_t ws_size,
                              hipStream_t stream) {
  (void)in_sizes; (void)n_in; (void)out_size; (void)d_ws; (void)ws_size;
  const float* Q = (const float*)d_in[0];
  const float* K = (const float*)d_in[1];
  const float* V = (const float*)d_in[2];
  const unsigned char* qm = (const unsigned char*)d_in[3];
  const unsigned char* km = (const unsigned char*)d_in[4];
  float* O = (float*)d_out;

  dim3 grid(L_ / QT, B_);   // 64 x 8 blocks
  dim3 block(NTHREADS);
  sdpa_flash_f16wmma<<<grid, block, sizeof(SMem), stream>>>(Q, K, V, qm, km, O);
}
